// MaskedAttentionOperation_15298673509078
// MI455X (gfx1250) — matrix operand order repaired and checked by reading, not compiled, not run on hardware
//
#include <hip/hip_runtime.h>
#include <hip/hip_bf16.h>
#include <math.h>

// ---------------------------------------------------------------------------
// MaskedAttention: sim=Q^T K (per b,r) -> global LN over [L,L] -> diag mask
// -> softmax -> AV -> batch LN -> exact GELU.   N=8 R=8 C=64 L=1024.
// Strategy (MI455X): bf16 WMMA, recompute QK instead of materializing sim,
// flash-style fused softmax+AV, fp32 accumulation throughout.
// ---------------------------------------------------------------------------

typedef __attribute__((ext_vector_type(16))) __bf16 v16bf;
typedef __attribute__((ext_vector_type(8)))  float  v8f;

#define LDIM 1024
#define CDIM 64
#define LDK  72          // LDS row stride (bf16 elems) for every tile

__device__ __forceinline__ v8f wmma_bf16(v16bf a, v16bf b, v8f c) {
  return __builtin_amdgcn_wmma_f32_16x16x32_bf16(
      /*neg_a=*/false, a, /*neg_b=*/false, b,
      /*c_mod=*/(short)0, c, /*reuse_a=*/false, /*reuse_b=*/false);
}

// A fragment (16x32 bf16, M x K): lane = M row; VGPR group/half select K pairs.
__device__ __forceinline__ v16bf load_frag_a(const __bf16* base, int lane) {
  const int m = lane & 15, half = lane >> 4;
  const __bf16* p = base + m * LDK;
  v16bf f;
#pragma unroll
  for (int g = 0; g < 2; ++g) {
    const int kb = half * 16 + g * 8;
#pragma unroll
    for (int i = 0; i < 8; ++i) f[g * 8 + i] = p[kb + i];
  }
  return f;
}

// B fragment (32x16 bf16, K x N) from LDS stored as [n][k]: lane = N col,
// halves split K 0..15 / 16..31; 16 contiguous bf16 per lane.
__device__ __forceinline__ v16bf load_frag_b(const __bf16* base, int lane) {
  const int n = lane & 15, half = lane >> 4;
  const __bf16* p = base + n * LDK + half * 16;
  v16bf f;
#pragma unroll
  for (int i = 0; i < 16; ++i) f[i] = p[i];
  return f;
}

__device__ __forceinline__ float wave_reduce(float v) {
#pragma unroll
  for (int off = 16; off > 0; off >>= 1) v += __shfl_xor(v, off, 32);
  return v;
}

// Stage [64][NCOLS] block of a [64][1024] fp32 tensor, TRANSPOSED, into LDS
// as bf16 [col][c] with row stride LDK. Coalesced float4 reads.
template <int NCOLS>
__device__ __forceinline__ void stage_T(const float* __restrict__ src, int col0,
                                        __bf16* dst, int tid) {
  constexpr int Q = NCOLS >> 2;
  for (int idx = tid; idx < CDIM * Q; idx += 256) {
    const int c = idx / Q, q4 = idx % Q;
    const float4 w = *reinterpret_cast<const float4*>(src + c * LDIM + col0 + (q4 << 2));
    const int lb = q4 << 2;
    dst[(lb + 0) * LDK + c] = (__bf16)w.x;
    dst[(lb + 1) * LDK + c] = (__bf16)w.y;
    dst[(lb + 2) * LDK + c] = (__bf16)w.z;
    dst[(lb + 3) * LDK + c] = (__bf16)w.w;
  }
}

// Stage [64][NM] block of V (no transpose: rows d, cols m contiguous) to bf16.
template <int NM>
__device__ __forceinline__ void stage_rows(const float* __restrict__ src, int m0,
                                           __bf16* dst, int tid) {
  constexpr int Q = NM >> 2;
  for (int idx = tid; idx < CDIM * Q; idx += 256) {
    const int d = idx / Q, q4 = idx % Q;
    const float4 w = *reinterpret_cast<const float4*>(src + d * LDIM + m0 + (q4 << 2));
    __bf16* p = dst + d * LDK + (q4 << 2);
    p[0] = (__bf16)w.x; p[1] = (__bf16)w.y; p[2] = (__bf16)w.z; p[3] = (__bf16)w.w;
  }
}

// ---------------------------------------------------------------------------
// Kernel 1: per-(b,r) sum / sumsq of sim = Q^T K, 128x128 tile per block.
// ---------------------------------------------------------------------------
__global__ __launch_bounds__(256) void k_sim_stats(const float* __restrict__ q,
                                                   const float* __restrict__ k,
                                                   float* __restrict__ stats1) {
  __shared__ __align__(16) unsigned char smem[2 * 18432 + 64];
  __bf16* Qs = (__bf16*)smem;
  __bf16* Ks = (__bf16*)(smem + 18432);
  float*  red = (float*)(smem + 36864);

  const int tid = threadIdx.x;
  const int p  = blockIdx.x >> 6;          // (b,r) pair, 0..63
  const int t  = blockIdx.x & 63;          // 8x8 tiles of 128x128
  const int tl = (t >> 3) << 7;
  const int tm = (t & 7) << 7;
  const float* qp = q + (size_t)p * CDIM * LDIM;
  const float* kp = k + (size_t)p * CDIM * LDIM;

  stage_T<128>(qp, tl, Qs, tid);
  stage_T<128>(kp, tm, Ks, tid);
  __syncthreads();

  const int wave = tid >> 5, lane = tid & 31;
  const __bf16* arow = Qs + (wave * 16) * LDK;
  const v16bf a0 = load_frag_a(arow, lane);
  const v16bf a1 = load_frag_a(arow + 32, lane);

  float lsum = 0.f, lsq = 0.f;
#pragma unroll
  for (int ns = 0; ns < 8; ++ns) {
    const __bf16* brow = Ks + (ns * 16) * LDK;
    const v16bf b0 = load_frag_b(brow, lane);
    const v16bf b1 = load_frag_b(brow + 32, lane);
    v8f acc = {0.f, 0.f, 0.f, 0.f, 0.f, 0.f, 0.f, 0.f};
    acc = wmma_bf16(a0, b0, acc);
    acc = wmma_bf16(a1, b1, acc);
#pragma unroll
    for (int j = 0; j < 8; ++j) { const float x = acc[j]; lsum += x; lsq += x * x; }
  }
  lsum = wave_reduce(lsum);
  lsq  = wave_reduce(lsq);
  if (lane == 0) { red[wave * 2] = lsum; red[wave * 2 + 1] = lsq; }
  __syncthreads();
  if (tid == 0) {
    float s = 0.f, ss = 0.f;
    for (int w = 0; w < 8; ++w) { s += red[w * 2]; ss += red[w * 2 + 1]; }
    atomicAdd(&stats1[p * 2 + 0], s);
    atomicAdd(&stats1[p * 2 + 1], ss);
  }
}

// ---------------------------------------------------------------------------
// Kernel 2: fused  LN(sim) -> diag mask -> exp -> rowsum -> O = P * V^T.
// Block = one (b,r) pair x 128 rows of l. Iterates 16 m-tiles of 64.
// ---------------------------------------------------------------------------
__global__ __launch_bounds__(256) void k_attn_av(const float* __restrict__ q,
                                                 const float* __restrict__ k,
                                                 const float* __restrict__ v,
                                                 const float* __restrict__ stats1,
                                                 float* __restrict__ obuf,
                                                 float* __restrict__ stats2) {
  // LDS carve-up (55872 B): Qs 18432 | Ks 9216 | Vs 9216 | Ps 18432 | rowsum 512 | red 64
  // Ofin (64x129 fp32 = 33024 B) aliases Ks/Vs/Ps after the last GEMM.
  __shared__ __align__(16) unsigned char smem[55872];
  __bf16* Qs    = (__bf16*)smem;
  __bf16* Ks    = (__bf16*)(smem + 18432);
  __bf16* Vs    = (__bf16*)(smem + 27648);
  __bf16* Ps    = (__bf16*)(smem + 36864);
  float*  rowsum = (float*)(smem + 55296);
  float*  red    = (float*)(smem + 55808);
  float*  Ofin   = (float*)(smem + 18432);

  const int tid = threadIdx.x;
  const int p  = blockIdx.x >> 3;
  const int tl = (blockIdx.x & 7) << 7;
  const int b = p >> 3, r = p & 7;
  const float* qp = q + (size_t)p * CDIM * LDIM;
  const float* kp = k + (size_t)p * CDIM * LDIM;
  const float* vp = v + (size_t)p * CDIM * LDIM;

  const float s1 = stats1[2 * p], s2 = stats1[2 * p + 1];
  const float invn = 1.0f / 1048576.0f;
  const float mean = s1 * invn;
  const float var  = fmaxf(s2 * invn - mean * mean, 0.f);
  const float rstd = rsqrtf(var + 1e-5f);

  stage_T<128>(qp, tl, Qs, tid);
  if (tid < 128) rowsum[tid] = 0.f;
  __syncthreads();

  const int wave = tid >> 5, lane = tid & 31;
  const int Ncol = lane & 15;
  const int hrow = wave * 16 + ((lane >> 4) << 3);  // first of 8 rows this lane owns

  const v16bf a0 = load_frag_a(Qs + (wave * 16) * LDK, lane);
  const v16bf a1 = load_frag_a(Qs + (wave * 16) * LDK + 32, lane);

  v8f o[4];
#pragma unroll
  for (int ds = 0; ds < 4; ++ds) o[ds] = (v8f){0.f, 0.f, 0.f, 0.f, 0.f, 0.f, 0.f, 0.f};

  for (int mi = 0; mi < 16; ++mi) {
    const int tm = mi << 6;
    stage_T<64>(kp, tm, Ks, tid);
    stage_rows<64>(vp, tm, Vs, tid);
    if (mi + 1 < 16) {  // global_prefetch_b8 next K/V tiles
      __builtin_prefetch(kp + ((tid >> 2) * LDIM + tm + 64), 0, 1);
      __builtin_prefetch(vp + ((tid >> 2) * LDIM + tm + 64), 0, 1);
    }
    __syncthreads();

    // ---- S tile: 16 rows x 64 m, per wave; LN + mask + exp, park in Ps ----
#pragma unroll
    for (int ns = 0; ns < 4; ++ns) {
      const __bf16* brow = Ks + (ns * 16) * LDK;
      const v16bf b0 = load_frag_b(brow, lane);
      const v16bf b1 = load_frag_b(brow + 32, lane);
      v8f acc = {0.f, 0.f, 0.f, 0.f, 0.f, 0.f, 0.f, 0.f};
      acc = wmma_bf16(a0, b0, acc);
      acc = wmma_bf16(a1, b1, acc);
      const int mg = tm + ns * 16 + Ncol;
#pragma unroll
      for (int j = 0; j < 8; ++j) {
        float x = __expf((acc[j] - mean) * rstd);
        if (tl + hrow + j == mg) x = 0.f;  // exp(-1e9) == 0
        Ps[(hrow + j) * LDK + ns * 16 + Ncol] = (__bf16)x;
      }
    }
    __syncthreads();

    // ---- row sums of P (accumulated across m-tiles) ----
    if (tid < 128) {
      float s = 0.f;
      const __bf16* pr = Ps + tid * LDK;
#pragma unroll
      for (int m = 0; m < 64; ++m) s += (float)pr[m];
      rowsum[tid] += s;
    }

    // ---- O += P (128x64) x V^T (64x64), accumulator lives in VGPRs ----
#pragma unroll
    for (int ks = 0; ks < 2; ++ks) {
      const v16bf ap = load_frag_a(Ps + (wave * 16) * LDK + ks * 32, lane);
#pragma unroll
      for (int ds = 0; ds < 4; ++ds) {
        const v16bf bv = load_frag_b(Vs + (ds * 16) * LDK + ks * 32, lane);
        o[ds] = wmma_bf16(ap, bv, o[ds]);
      }
    }
    __syncthreads();  // Ks/Vs/Ps reused next iteration
  }

  // ---- normalize by rowsum, stage [d][l] in LDS for coalesced store ----
#pragma unroll
  for (int ds = 0; ds < 4; ++ds) {
#pragma unroll
    for (int j = 0; j < 8; ++j) {
      const int lr = hrow + j;
      const float rs = rowsum[lr];
      const float val = (rs > 0.f) ? o[ds][j] / rs : 0.f;  // NaN-safe
      Ofin[(ds * 16 + Ncol) * 129 + lr] = val;
    }
  }
  __syncthreads();

  float bs = 0.f, bq = 0.f;
  for (int idx = tid; idx < 64 * 128; idx += 256) {
    const int d = idx >> 7, lj = idx & 127;
    const float x = Ofin[d * 129 + lj];
    obuf[(size_t)(b * 512 + r * 64 + d) * LDIM + tl + lj] = x;
    bs += x; bq += x * x;
  }
  bs = wave_reduce(bs);
  bq = wave_reduce(bq);
  if (lane == 0) { red[wave * 2] = bs; red[wave * 2 + 1] = bq; }
  __syncthreads();
  if (tid == 0) {
    float s = 0.f, ss = 0.f;
    for (int w = 0; w < 8; ++w) { s += red[w * 2]; ss += red[w * 2 + 1]; }
    atomicAdd(&stats2[b * 2 + 0], s);
    atomicAdd(&stats2[b * 2 + 1], ss);
  }
}

// ---------------------------------------------------------------------------
// Kernel 3: per-batch LayerNorm over [512,1024] + exact GELU.
// ---------------------------------------------------------------------------
__global__ __launch_bounds__(256) void k_ln_gelu(const float* __restrict__ obuf,
                                                 const float* __restrict__ stats2,
                                                 float* __restrict__ out) {
  const int i = (blockIdx.x * 256 + threadIdx.x) << 2;  // 4 elems/thread
  const int b = i >> 19;                                // 512*1024 per batch
  const float invn = 1.0f / 524288.0f;
  const float mean = stats2[b * 2] * invn;
  const float var  = fmaxf(stats2[b * 2 + 1] * invn - mean * mean, 0.f);
  const float rstd = rsqrtf(var + 1e-5f);
  float4 x = *reinterpret_cast<const float4*>(obuf + i);
  float4 g;
  {
    const float t = (x.x - mean) * rstd; g.x = 0.5f * t * (1.f + erff(t * 0.70710678f));
  }
  {
    const float t = (x.y - mean) * rstd; g.y = 0.5f * t * (1.f + erff(t * 0.70710678f));
  }
  {
    const float t = (x.z - mean) * rstd; g.z = 0.5f * t * (1.f + erff(t * 0.70710678f));
  }
  {
    const float t = (x.w - mean) * rstd; g.w = 0.5f * t * (1.f + erff(t * 0.70710678f));
  }
  *reinterpret_cast<float4*>(out + i) = g;
}

// ---------------------------------------------------------------------------
extern "C" void kernel_launch(void* const* d_in, const int* in_sizes, int n_in,
                              void* d_out, int out_size, void* d_ws, size_t ws_size,
                              hipStream_t stream) {
  const float* q = (const float*)d_in[0];
  const float* k = (const float*)d_in[1];
  const float* v = (const float*)d_in[2];
  float* out = (float*)d_out;

  float* stats1 = (float*)d_ws;          // 128 floats: per-(b,r) sum/sumsq
  float* stats2 = stats1 + 128;          // 16 floats : per-batch sum/sumsq
  float* obuf   = stats1 + 256;          // 8*512*1024 floats pre-LN output

  hipMemsetAsync(d_ws, 0, 256 * sizeof(float), stream);
  k_sim_stats<<<dim3(64 * 64), dim3(256), 0, stream>>>(q, k, stats1);
  k_attn_av<<<dim3(64 * 8), dim3(256), 0, stream>>>(q, k, v, stats1, obuf, stats2);
  k_ln_gelu<<<dim3(4096), dim3(256), 0, stream>>>(obuf, stats2, out);
}
